// Create_19301583028978
// MI455X (gfx1250) — matrix operand order repaired and checked by reading, not compiled, not run on hardware
//
#include <hip/hip_runtime.h>

typedef __attribute__((ext_vector_type(16))) __bf16 v16bf;
typedef __attribute__((ext_vector_type(8)))  float  v8f;

struct U32x8 { uint4 lo; uint4 hi; };

__device__ __forceinline__ ushort f2bf(float f) {
    uint u = __float_as_uint(f);
    u += 0x7FFFu + ((u >> 16) & 1u);        // round-to-nearest-even
    return (ushort)(u >> 16);
}

#define BN_EPS 1e-5f

// ---------------------------------------------------------------------------
// WMMA GEMM:  C[M,N] = epilogue( A[M,K] @ B[K,N] ), all operands bf16.
//   AMODE 1: A bf16 row-major (ld = K), per-lane direct global b128 fragments
//   AMODE 2: A gathered: k<512 -> dfb[dix[m]][k], else fsb[six[m]][k-512]
//   EPI   0: C = f32 plain;  EPI 1: C = bf16, lrelu_0.01(acc*scale+shift)
// Tile: BM=128, BN=64, BK=32, 256 thr (8 waves), wave -> 16-row strip.
// B tile: global_load_async_to_lds_b128 double-buffered (ASYNCcnt);
// A fragment software-pipelined in registers (k+1 load overlaps k compute).
// ---------------------------------------------------------------------------
template<int AMODE, int EPI>
__global__ __launch_bounds__(256)
void gemm_wmma(const ushort* __restrict__ A,
               const ushort* __restrict__ dfb, const ushort* __restrict__ fsb,
               const int* __restrict__ dix, const int* __restrict__ six,
               const ushort* __restrict__ B, void* __restrict__ Cp,
               const float* __restrict__ bnb, const float* __restrict__ bng,
               const float* __restrict__ bnbe, const float* __restrict__ bnm,
               const float* __restrict__ bnv,
               int M, int N, int K)
{
    __shared__ ushort lB[2][32 * 72];   // 2 x (32 rows x (64+8) bf16), 9216 B

    const int tid  = threadIdx.x;
    const int wave = tid >> 5;
    const int lane = tid & 31;
    const int h    = lane >> 4;
    const int lm   = lane & 15;
    const int n0   = blockIdx.x * 64;
    const int m0   = blockIdx.y * 128;

    // ---- async B staging: thread owns one 16B chunk: row=tid>>3, 8 bf16 ----
    const int brow   = tid >> 3;          // 0..31
    const int bchunk = (tid & 7) * 8;     // bf16 element within row
    const uint lds0 = (uint)(size_t)&lB[0][brow * 72 + bchunk];
    const uint lds1 = (uint)(size_t)&lB[1][brow * 72 + bchunk];

    auto issueB = [&](int buf, int k0) {
        uint lds  = buf ? lds1 : lds0;
        int  goff = (((k0 + brow) * N) + n0 + bchunk) * 2;   // byte offset
        asm volatile("global_load_async_to_lds_b128 %0, %1, %2"
                     :: "v"(lds), "v"(goff), "s"(B) : "memory");
    };

    const int arow = m0 + wave * 16 + lm;      // A row this lane fetches
    const ushort* arp = nullptr;
    int gdi = 0, gsi = 0;
    if (AMODE == 1) arp = A + (size_t)arow * K;
    if (AMODE == 2) { gdi = dix[arow]; gsi = six[arow]; }

    auto loadA = [&](int k0) {
        const ushort* abase;
        if (AMODE == 2)
            abase = (k0 < 512) ? (dfb + (size_t)gdi * 512 + k0)
                               : (fsb + (size_t)gsi * 256 + (k0 - 512));
        else
            abase = arp + k0;
        U32x8 au;
        au.lo = *(const uint4*)(abase + 8 * h);
        au.hi = *(const uint4*)(abase + 16 + 8 * h);
        return au;
    };

    v8f acc[4] = {};
    auto wmma4 = [&](const U32x8& au, const ushort* lbb) {
        v16bf a = __builtin_bit_cast(v16bf, au);
#pragma unroll
        for (int j = 0; j < 4; j++) {
            const ushort* bb = lbb + j * 16 + lm;     // N=lm, 16 K vals (table order)
            v16bf b;
#pragma unroll
            for (int s = 0; s < 16; s++)
                b[s] = __builtin_bit_cast(__bf16, bb[((s & 8) * 2 + 8 * h + (s & 7)) * 72]);
            acc[j] = __builtin_amdgcn_wmma_f32_16x16x32_bf16(
                false, a, false, b, (short)0, acc[j], false, false);
        }
    };

    const int nk = K >> 5;
    issueB(0, 0);                              // prologue: B tile 0 -> buf 0
    U32x8 au = loadA(0);                       // prologue: A frag 0

    for (int kb = 0; kb < nk - 1; kb++) {      // steady state: branch-free body
        const int cur = kb & 1;
        issueB(cur ^ 1, (kb + 1) * 32);        // prefetch next B tile
        asm volatile("s_wait_asynccnt 0x1" ::: "memory");   // tile kb in LDS
        __syncthreads();
        U32x8 au_next = loadA((kb + 1) * 32);  // overlaps with WMMAs below
        wmma4(au, lB[cur]);
        __syncthreads();
        au = au_next;
    }
    // epilogue iteration
    asm volatile("s_wait_asynccnt 0x0" ::: "memory");
    __syncthreads();
    wmma4(au, lB[(nk - 1) & 1]);

    // ---- epilogue: C rows m=i+8h, col = lane&15 ----
    const int row0 = m0 + wave * 16 + 8 * h;
#pragma unroll
    for (int j = 0; j < 4; j++) {
        const int col = n0 + j * 16 + lm;
        if (EPI == 1) {
            float s = bng[col] * rsqrtf(bnv[col] + BN_EPS);
            float t = (bnb[col] - bnm[col]) * s + bnbe[col];
            ushort* out = (ushort*)Cp;
#pragma unroll
            for (int i = 0; i < 8; i++) {
                float v = acc[j][i] * s + t;
                v = v > 0.f ? v : 0.01f * v;
                out[(size_t)(row0 + i) * N + col] = f2bf(v);
            }
        } else {
            float* out = (float*)Cp;
#pragma unroll
            for (int i = 0; i < 8; i++)
                out[(size_t)(row0 + i) * N + col] = acc[j][i];
        }
    }
}

// ---------------------------------------------------------------------------
// Small kernels
// ---------------------------------------------------------------------------
__global__ void cvt_bf16(const float* __restrict__ in, ushort* __restrict__ out, int n4) {
    int i = blockIdx.x * blockDim.x + threadIdx.x;
    if (i >= n4) return;
    float4 v = ((const float4*)in)[i];
    ushort4 s; s.x = f2bf(v.x); s.y = f2bf(v.y); s.z = f2bf(v.z); s.w = f2bf(v.w);
    ((ushort4*)out)[i] = s;
}

__global__ void zero_f32(float* p, int n) {
    int i = blockIdx.x * blockDim.x + threadIdx.x;
    if (i < n) p[i] = 0.f;
}

__global__ void rowdot_f32(const float* __restrict__ X, const float* __restrict__ w,
                           float* __restrict__ out, int rows) {
    int wv = (blockIdx.x * blockDim.x + threadIdx.x) >> 5;
    int lane = threadIdx.x & 31;
    if (wv >= rows) return;
    const float* xp = X + (size_t)wv * 256;
    float s = 0.f;
#pragma unroll
    for (int i = 0; i < 8; i++) s += xp[i * 32 + lane] * w[i * 32 + lane];
    for (int off = 16; off; off >>= 1) s += __shfl_down(s, off, 32);
    if (lane == 0) out[wv] = s;
}

__global__ void head_i5(const ushort* __restrict__ X, const float* __restrict__ w,
                        const float* __restrict__ b, float* __restrict__ out, int rows) {
    int wv = (blockIdx.x * blockDim.x + threadIdx.x) >> 5;
    int lane = threadIdx.x & 31;
    if (wv >= rows) return;
    const ushort* xp = X + (size_t)wv * 256 + lane * 8;
    const float* wp = w + lane * 8;
    uint4 q = *(const uint4*)xp;
    uint u[4] = {q.x, q.y, q.z, q.w};
    float s = 0.f;
#pragma unroll
    for (int i = 0; i < 4; i++) {
        s += __uint_as_float(u[i] << 16) * wp[2 * i];
        s += __uint_as_float(u[i] & 0xFFFF0000u) * wp[2 * i + 1];
    }
    for (int off = 16; off; off >>= 1) s += __shfl_down(s, off, 32);
    if (lane == 0) out[wv] = 1.f / (1.f + __expf(-(s + b[0])));
}

__device__ __forceinline__ uint flip_f(float f) {
    uint u = __float_as_uint(f);
    return (u & 0x80000000u) ? ~u : (u | 0x80000000u);
}
__device__ __forceinline__ float unflip_f(uint u) {
    return __uint_as_float((u & 0x80000000u) ? (u ^ 0x80000000u) : ~u);
}

__global__ void edge_pass1(const int* __restrict__ src, const int* __restrict__ dst,
                           const float* __restrict__ al, const float* __restrict__ ar,
                           float* __restrict__ ebuf, uint* __restrict__ emax, int ne) {
    int e = blockIdx.x * blockDim.x + threadIdx.x;
    if (e >= ne) return;
    int s = src[e], d = dst[e];
    float v = al[s] + ar[d];
    v = v > 0.f ? v : 0.2f * v;          // leaky_relu(0.2)
    ebuf[e] = v;
    atomicMax(&emax[d], flip_f(v));
}

__global__ void edge_pass2(const int* __restrict__ dst, float* __restrict__ ebuf,
                           const uint* __restrict__ emax, float* __restrict__ den, int ne) {
    int e = blockIdx.x * blockDim.x + threadIdx.x;
    if (e >= ne) return;
    int d = dst[e];
    float ex = __expf(ebuf[e] - unflip_f(emax[d]));
    ebuf[e] = ex;
    atomicAdd(&den[d], ex);
}

__global__ void edge_agg(const int* __restrict__ src, const int* __restrict__ dst,
                         const float* __restrict__ ebuf, const float* __restrict__ den,
                         const float* __restrict__ xl, float* __restrict__ agg, int ne) {
    int e = (blockIdx.x * blockDim.x + threadIdx.x) >> 5;
    int lane = threadIdx.x & 31;
    if (e >= ne) return;
    int s = src[e], d = dst[e];
    float w = ebuf[e] / den[d];
    const float* xp = xl + (size_t)s * 256;
    float* ap = agg + (size_t)d * 256;
#pragma unroll
    for (int i = 0; i < 8; i++) {
        int c = i * 32 + lane;
        atomicAdd(&ap[c], xp[c] * w);
    }
}

__global__ void fse_kernel(const float* __restrict__ agg, const float* __restrict__ xr,
                           const float* __restrict__ bias, float* __restrict__ fse_out,
                           ushort* __restrict__ fsb, int total) {
    int i = blockIdx.x * blockDim.x + threadIdx.x;
    if (i >= total) return;
    float x = agg[i] + xr[i] + bias[i & 255];
    float f = 1.f / (1.f + __expf(-x));
    fse_out[i] = f;
    fsb[i] = f2bf(f);
}

// ---------------------------------------------------------------------------
extern "C" void kernel_launch(void* const* d_in, const int* in_sizes, int n_in,
                              void* d_out, int out_size, void* d_ws, size_t ws_size,
                              hipStream_t stream) {
    const int ND = 4096, NS = 8192, NE = 262144, NP = 131072;

    const int*   ei   = (const int*)d_in[0];
    const float* dstr = (const float*)d_in[1];
    const float* dexp = (const float*)d_in[2];
    const float* sest = (const float*)d_in[3];
    const int*   dix  = (const int*)d_in[4];
    const int*   six  = (const int*)d_in[5];
    const int* esrc = ei;
    const int* edst = ei + NE;

    const float* d1W = (const float*)d_in[6];
    const float* d1b = (const float*)d_in[7],  *d1g = (const float*)d_in[8];
    const float* d1be = (const float*)d_in[9], *d1m = (const float*)d_in[10], *d1v = (const float*)d_in[11];
    const float* d2W = (const float*)d_in[12];
    const float* d2b = (const float*)d_in[13], *d2g = (const float*)d_in[14];
    const float* d2be = (const float*)d_in[15], *d2m = (const float*)d_in[16], *d2v = (const float*)d_in[17];
    const float* gWl = (const float*)d_in[18];
    const float* gWr = (const float*)d_in[19];
    const float* gal = (const float*)d_in[20];
    const float* gar = (const float*)d_in[21];
    const float* gbias = (const float*)d_in[22];
    const float* i1W = (const float*)d_in[23];
    const float* i1b = (const float*)d_in[24], *i1g = (const float*)d_in[25];
    const float* i1be = (const float*)d_in[26], *i1m = (const float*)d_in[27], *i1v = (const float*)d_in[28];
    const float* i2W = (const float*)d_in[29];
    const float* i2b = (const float*)d_in[30], *i2g = (const float*)d_in[31];
    const float* i2be = (const float*)d_in[32], *i2m = (const float*)d_in[33], *i2v = (const float*)d_in[34];
    const float* i3W = (const float*)d_in[35];
    const float* i3b = (const float*)d_in[36], *i3g = (const float*)d_in[37];
    const float* i3be = (const float*)d_in[38], *i3m = (const float*)d_in[39], *i3v = (const float*)d_in[40];
    const float* i5W = (const float*)d_in[41];
    const float* i5b = (const float*)d_in[42];

    float* adj     = (float*)d_out;
    float* fse_out = adj + NP;

    // ---- workspace layout ----
    char* ws = (char*)d_ws;
    const size_t OFF_H1 = 0;                                  // bf16 [NP,1024]
    const size_t OFF_H3 = 0;                                  // bf16 [NP,256], aliases dead h1
    const size_t OFF_H2 = (size_t)NP * 1024 * 2;              // bf16 [NP,512]
    size_t o = OFF_H2 + (size_t)NP * 512 * 2;
    const size_t OFF_H1D = o;  o += (size_t)ND * 1024 * 2;
    const size_t OFF_DFB = o;  o += (size_t)ND * 512 * 2;
    const size_t OFF_FSB = o;  o += (size_t)NS * 256 * 2;
    const size_t OFF_XL  = o;  o += (size_t)ND * 256 * 4;
    const size_t OFF_XR  = o;  o += (size_t)NS * 256 * 4;
    const size_t OFF_AL  = o;  o += (size_t)ND * 4;
    const size_t OFF_AR  = o;  o += (size_t)NS * 4;
    const size_t OFF_EB  = o;  o += (size_t)NE * 4;
    const size_t OFF_EMX = o;  o += (size_t)NS * 4;
    const size_t OFF_DEN = o;  o += (size_t)NS * 4;
    const size_t OFF_AGG = o;  o += (size_t)NS * 256 * 4;
    const size_t OFF_DSTRB = o; o += (size_t)ND * 2048 * 2;
    const size_t OFF_DEXPB = o; o += (size_t)ND * 1024 * 2;
    const size_t OFF_SESTB = o; o += (size_t)NS * 1024 * 2;
    const size_t OFF_W1B   = o; o += (size_t)2048 * 1024 * 2;
    const size_t OFF_W2B   = o; o += (size_t)1024 * 512 * 2;
    const size_t OFF_WLB   = o; o += (size_t)1024 * 256 * 2;
    const size_t OFF_WRB   = o; o += (size_t)1024 * 256 * 2;
    const size_t OFF_WI1B  = o; o += (size_t)768 * 1024 * 2;
    const size_t OFF_WI2B  = o; o += (size_t)1024 * 512 * 2;
    const size_t OFF_WI3B  = o; o += (size_t)512 * 256 * 2;

    ushort* h1   = (ushort*)(ws + OFF_H1);
    ushort* h2   = (ushort*)(ws + OFF_H2);
    ushort* h3   = (ushort*)(ws + OFF_H3);
    ushort* h1d  = (ushort*)(ws + OFF_H1D);
    ushort* dfb  = (ushort*)(ws + OFF_DFB);
    ushort* fsb  = (ushort*)(ws + OFF_FSB);
    float*  xl   = (float*)(ws + OFF_XL);
    float*  xr   = (float*)(ws + OFF_XR);
    float*  al   = (float*)(ws + OFF_AL);
    float*  ar   = (float*)(ws + OFF_AR);
    float*  eb   = (float*)(ws + OFF_EB);
    uint*   emx  = (uint*)(ws + OFF_EMX);
    float*  den  = (float*)(ws + OFF_DEN);
    float*  agg  = (float*)(ws + OFF_AGG);
    ushort* dstrb = (ushort*)(ws + OFF_DSTRB);
    ushort* dexpb = (ushort*)(ws + OFF_DEXPB);
    ushort* sestb = (ushort*)(ws + OFF_SESTB);
    ushort* w1b  = (ushort*)(ws + OFF_W1B);
    ushort* w2b  = (ushort*)(ws + OFF_W2B);
    ushort* wlb  = (ushort*)(ws + OFF_WLB);
    ushort* wrb  = (ushort*)(ws + OFF_WRB);
    ushort* wi1b = (ushort*)(ws + OFF_WI1B);
    ushort* wi2b = (ushort*)(ws + OFF_WI2B);
    ushort* wi3b = (ushort*)(ws + OFF_WI3B);

    // ---- one-time (per launch) f32 -> bf16 conversions ----
    auto cvt = [&](const float* in, ushort* out, size_t n) {
        int n4 = (int)(n / 4);
        cvt_bf16<<<(n4 + 255) / 256, 256, 0, stream>>>(in, out, n4);
    };
    cvt(dstr, dstrb, (size_t)ND * 2048);
    cvt(dexp, dexpb, (size_t)ND * 1024);
    cvt(sest, sestb, (size_t)NS * 1024);
    cvt(d1W, w1b, (size_t)2048 * 1024);
    cvt(d2W, w2b, (size_t)1024 * 512);
    cvt(gWl, wlb, (size_t)1024 * 256);
    cvt(gWr, wrb, (size_t)1024 * 256);
    cvt(i1W, wi1b, (size_t)768 * 1024);
    cvt(i2W, wi2b, (size_t)1024 * 512);
    cvt(i3W, wi3b, (size_t)512 * 256);

    // ---- zero accumulators ----
    zero_f32<<<(NS * 256 + 255) / 256, 256, 0, stream>>>(agg, NS * 256);
    zero_f32<<<(NS + 255) / 256, 256, 0, stream>>>(den, NS);
    zero_f32<<<(NS + 255) / 256, 256, 0, stream>>>((float*)emx, NS);

    // ---- Drugin ----
    gemm_wmma<1, 1><<<dim3(1024 / 64, ND / 128), 256, 0, stream>>>(
        dstrb, nullptr, nullptr, nullptr, nullptr, w1b, h1d,
        d1b, d1g, d1be, d1m, d1v, ND, 1024, 2048);
    gemm_wmma<1, 1><<<dim3(512 / 64, ND / 128), 256, 0, stream>>>(
        h1d, nullptr, nullptr, nullptr, nullptr, w2b, dfb,
        d2b, d2g, d2be, d2m, d2v, ND, 512, 1024);

    // ---- GAT node projections ----
    gemm_wmma<1, 0><<<dim3(256 / 64, ND / 128), 256, 0, stream>>>(
        dexpb, nullptr, nullptr, nullptr, nullptr, wlb, xl,
        nullptr, nullptr, nullptr, nullptr, nullptr, ND, 256, 1024);
    gemm_wmma<1, 0><<<dim3(256 / 64, NS / 128), 256, 0, stream>>>(
        sestb, nullptr, nullptr, nullptr, nullptr, wrb, xr,
        nullptr, nullptr, nullptr, nullptr, nullptr, NS, 256, 1024);

    rowdot_f32<<<ND * 32 / 256, 256, 0, stream>>>(xl, gal, al, ND);
    rowdot_f32<<<NS * 32 / 256, 256, 0, stream>>>(xr, gar, ar, NS);

    // ---- GAT edge softmax + aggregation ----
    edge_pass1<<<NE / 256, 256, 0, stream>>>(esrc, edst, al, ar, eb, emx, NE);
    edge_pass2<<<NE / 256, 256, 0, stream>>>(edst, eb, emx, den, NE);
    edge_agg<<<NE * 32 / 256, 256, 0, stream>>>(esrc, edst, eb, den, xl, agg, NE);
    fse_kernel<<<NS * 256 / 256, 256, 0, stream>>>(agg, xr, gbias, fse_out, fsb, NS * 256);

    // ---- Interact MLP (gathered-concat A fused into i1) ----
    gemm_wmma<2, 1><<<dim3(1024 / 64, NP / 128), 256, 0, stream>>>(
        nullptr, dfb, fsb, dix, six, wi1b, h1,
        i1b, i1g, i1be, i1m, i1v, NP, 1024, 768);
    gemm_wmma<1, 1><<<dim3(512 / 64, NP / 128), 256, 0, stream>>>(
        h1, nullptr, nullptr, nullptr, nullptr, wi2b, h2,
        i2b, i2g, i2be, i2m, i2v, NP, 512, 1024);
    gemm_wmma<1, 1><<<dim3(256 / 64, NP / 128), 256, 0, stream>>>(
        h2, nullptr, nullptr, nullptr, nullptr, wi3b, h3,
        i3b, i3g, i3be, i3m, i3v, NP, 256, 512);

    head_i5<<<NP * 32 / 256, 256, 0, stream>>>(h3, i5W, i5b, adj, NP);
}
